// GCN_39049842655815
// MI455X (gfx1250) — compile-verified, matrix operand order checked by reading, not run on hardware
//
#include <hip/hip_runtime.h>
#include <hip/hip_bf16.h>

// ---------------- problem constants ----------------
#define GN    16384   // num nodes
#define GDIN  512     // input dim
#define GDH   128     // hidden dim
#define GDOUT 40      // output classes
#define GDPAD 64      // padded output dim for WMMA tiles
#define GNSUP 1024

typedef __attribute__((ext_vector_type(16))) __bf16 v16bf;
typedef __attribute__((ext_vector_type(8)))  float  v8f;

// bf16 storage type (avoid relying on __bf16 arithmetic; only the WMMA
// operand is typed as v16bf via a bit-pun).
typedef unsigned short bf16_t;

__device__ __forceinline__ bf16_t f2bf(float f) {
    union { float f; unsigned int u; } v; v.f = f;
    unsigned int u = v.u;
    unsigned int r = (u + 0x7FFFu + ((u >> 16) & 1u)) >> 16;  // RNE
    return (bf16_t)r;
}

__device__ __forceinline__ unsigned int pack2bf(float a, float b) {
    return (unsigned int)f2bf(a) | ((unsigned int)f2bf(b) << 16);
}

// Exact truncating pack for values already exactly representable in bf16
// (adjacency entries are {0.0, 1.0, 2.0}): one v_perm_b32 grabs the high
// 16 bits of both floats. dst = [b[31:16] : a[31:16]].
__device__ __forceinline__ unsigned int pack2bf_exact(float a, float b) {
    union { float f; unsigned int u; } ua, ub;
    ua.f = a; ub.f = b;
    // v_perm_b32: src0 supplies bytes 7..4, src1 supplies bytes 3..0.
    // sel 0x07060302 -> {b.byte3, b.byte2, a.byte3, a.byte2}
    return __builtin_amdgcn_perm(ub.u, ua.u, 0x07060302u);
}

struct FragBits { uint4 lo; uint4 hi; };
union FragPun { FragBits b; v16bf v; };

// Load a 16x32 bf16 WMMA operand fragment from an LDS tile.
// ISA layout (16-bit A/B, wave32): lanes 0-15 -> row/col = lane, K chunks
// {0..7} and {16..23}; lanes 16-31 -> row/col = lane-16, K chunks {8..15}
// and {24..31}. Each chunk = 8 contiguous bf16 = one 16B ds_load_b128.
__device__ __forceinline__ v16bf load_frag(const bf16_t* base, int stride, int lane) {
    int half = lane >> 4;
    int r    = lane & 15;
    const bf16_t* p = base + r * stride + half * 8;
    FragPun u;
    u.b.lo = *(const uint4*)(p);
    u.b.hi = *(const uint4*)(p + 16);
    return u.v;
}

// ---------------- normalization kernels ----------------

__global__ __launch_bounds__(256) void rownorm_kernel(const float* __restrict__ adj,
                                                      float* __restrict__ row_inv) {
    int i   = blockIdx.x;
    int tid = threadIdx.x;
    const float* row = adj + (size_t)i * GN;
    float s = 0.f;
    for (int j = tid; j < GN; j += 256) {
        float v = row[j] + ((j == i) ? 1.0f : 0.0f);
        s += v * v;
    }
    __shared__ float red[256];
    red[tid] = s;
    __syncthreads();
    for (int off = 128; off > 0; off >>= 1) {
        if (tid < off) red[tid] += red[tid + off];
        __syncthreads();
    }
    if (tid == 0) {
        float t = red[0];
        row_inv[i] = (t > 0.f) ? rsqrtf(sqrtf(t)) : 0.f;
    }
}

__global__ __launch_bounds__(256) void colsq_init_kernel(float* __restrict__ colsq) {
    colsq[blockIdx.x * 256 + threadIdx.x] = 0.f;
}

__global__ __launch_bounds__(256) void colsq_part_kernel(const float* __restrict__ adj,
                                                         float* __restrict__ colsq) {
    int j  = blockIdx.x * 256 + threadIdx.x;
    int r0 = blockIdx.y * (GN / 32);
    int r1 = r0 + (GN / 32);
    float s = 0.f;
    for (int r = r0; r < r1; ++r) {
        float v = adj[(size_t)r * GN + j] + ((r == j) ? 1.0f : 0.0f);
        s += v * v;
    }
    atomicAdd(&colsq[j], s);
}

__global__ __launch_bounds__(256) void colinv_kernel(const float* __restrict__ colsq,
                                                     float* __restrict__ col_inv) {
    int j = blockIdx.x * 256 + threadIdx.x;
    float t = colsq[j];
    col_inv[j] = (t > 0.f) ? rsqrtf(sqrtf(t)) : 0.f;
}

// ---------------- double-buffered WMMA GEMM ----------------
// C[M x BN] = A[M x K] * B[K x BN]
// ASRC 0: A = f32 row-major [M x K] (optionally + I)
// ASRC 1: A = bf16 row-major [M x K]
// BSRC 0: B = f32 row-major [K x nrealB], zero-padded to BN cols (transposed in-kernel)
// BSRC 1: B = bf16 TRANSPOSED [BN x K] row-major (direct uint4 copy)
// EPI 0: bf16 transposed store: outT[col*ldo + row] = scale[row]*acc
// EPI 1: bf16 row-major store:  out[row*ldo + col] = relu(scale[row]*acc)
// EPI 2: f32 row-major store:   out[row*ncolsOut + col] = scale[row]*acc (col<ncolsOut)
template<int BM, int BN, int BK, int WGM, int WGN,
         bool ADD_DIAG, int ASRC, int BSRC, int EPI>
__global__ __launch_bounds__(256) void gemm_wmma_kernel(
        const void* __restrict__ Ap, int lda,
        const void* __restrict__ Bp, int ldb, int nrealB,
        const float* __restrict__ scale,
        void* __restrict__ outp, int ldo, int ncolsOut,
        int M, int K) {
    constexpr int SA  = BK + 8;            // LDS row stride (bf16): 72 -> 144B, 16B-multiple
    constexpr int WTM = BM / (WGM * 16);
    constexpr int WTN = BN / (WGN * 16);
    constexpr int KH  = BK / 32;           // WMMA K-steps per stage
    static_assert(WGM * WGN == 8, "8 waves");

    __shared__ alignas(16) bf16_t As[2][BM * SA];
    __shared__ alignas(16) bf16_t Bs[2][BN * SA];

    const int tid      = threadIdx.x;
    const int wave     = tid >> 5;
    const int lane     = tid & 31;
    const int waveM    = wave / WGN;
    const int waveN    = wave % WGN;
    const int blockRow = blockIdx.y * BM;
    const int rowBase  = waveM * (WTM * 16);
    const int colBase  = waveN * (WTN * 16);

    // staging registers
    constexpr int NA4 = (ASRC == 0) ? (BM * BK) / (4 * 256) : 1;   // float4 count
    constexpr int NAH = (ASRC == 1) ? (BM * BK) / (8 * 256) : 1;   // uint4 count
    constexpr int NB4 = (BSRC == 0) ? (BK * BN) / (4 * 256) : 1;   // float4 count
    constexpr int NBH = (BSRC == 1) ? (BN * BK) / (8 * 256) : 1;   // uint4 count
    float4 aReg4[NA4];
    uint4  aRegH[NAH];
    float4 bReg4[NB4];
    uint4  bRegH[NBH];

    v8f acc[WTM][WTN];
#pragma unroll
    for (int a = 0; a < WTM; ++a)
#pragma unroll
        for (int b = 0; b < WTN; ++b)
            acc[a][b] = v8f{0.f, 0.f, 0.f, 0.f, 0.f, 0.f, 0.f, 0.f};

    // ---- global -> registers (issue loads, no conversion) ----
    auto loadA = [&](int k0) {
        if (ASRC == 0) {
#pragma unroll
            for (int i = 0; i < NA4; ++i) {
                int idx = tid + i * 256;
                int row = idx / (BK / 4), c4 = idx % (BK / 4);
                aReg4[i] = *(const float4*)((const float*)Ap +
                               (size_t)(blockRow + row) * lda + k0 + c4 * 4);
            }
        } else {
#pragma unroll
            for (int i = 0; i < NAH; ++i) {
                int idx = tid + i * 256;
                int row = idx / (BK / 8), c8 = idx % (BK / 8);
                aRegH[i] = *(const uint4*)((const bf16_t*)Ap +
                               (size_t)(blockRow + row) * lda + k0 + c8 * 8);
            }
        }
    };
    auto loadB = [&](int k0) {
        if (BSRC == 0) {
#pragma unroll
            for (int i = 0; i < NB4; ++i) {
                int idx = tid + i * 256;
                int k = idx / (BN / 4), n4 = idx % (BN / 4);
                const float* src = (const float*)Bp + (size_t)(k0 + k) * ldb + n4 * 4;
                float4 v;
                if (n4 * 4 + 3 < nrealB) {
                    v = *(const float4*)src;
                } else {
                    v.x = (n4 * 4 + 0 < nrealB) ? src[0] : 0.f;
                    v.y = (n4 * 4 + 1 < nrealB) ? src[1] : 0.f;
                    v.z = (n4 * 4 + 2 < nrealB) ? src[2] : 0.f;
                    v.w = (n4 * 4 + 3 < nrealB) ? src[3] : 0.f;
                }
                bReg4[i] = v;
            }
        } else {
#pragma unroll
            for (int i = 0; i < NBH; ++i) {
                int idx = tid + i * 256;
                int n = idx / (BK / 8), c8 = idx % (BK / 8);
                bRegH[i] = *(const uint4*)((const bf16_t*)Bp +
                               (size_t)n * ldb + k0 + c8 * 8);
            }
        }
    };
    // ---- registers -> LDS (convert where needed) ----
    auto storeA = [&](int buf, int k0) {
        if (ASRC == 0) {
#pragma unroll
            for (int i = 0; i < NA4; ++i) {
                int idx = tid + i * 256;
                int row = idx / (BK / 4), c4 = idx % (BK / 4);
                int gr = blockRow + row, gc = k0 + c4 * 4;
                float4 fv = aReg4[i];
                uint2 p;
                if (ADD_DIAG) {
                    // adjacency path: entries are {0,1} (+1 on the diagonal),
                    // all exact in bf16 -> truncating v_perm pack, no rounding.
                    fv.x += (gr == gc + 0) ? 1.f : 0.f;
                    fv.y += (gr == gc + 1) ? 1.f : 0.f;
                    fv.z += (gr == gc + 2) ? 1.f : 0.f;
                    fv.w += (gr == gc + 3) ? 1.f : 0.f;
                    p.x = pack2bf_exact(fv.x, fv.y);
                    p.y = pack2bf_exact(fv.z, fv.w);
                } else {
                    p.x = pack2bf(fv.x, fv.y);
                    p.y = pack2bf(fv.z, fv.w);
                }
                *(uint2*)&As[buf][row * SA + c4 * 4] = p;
            }
        } else {
#pragma unroll
            for (int i = 0; i < NAH; ++i) {
                int idx = tid + i * 256;
                int row = idx / (BK / 8), c8 = idx % (BK / 8);
                *(uint4*)&As[buf][row * SA + c8 * 8] = aRegH[i];
            }
        }
    };
    auto storeB = [&](int buf) {
        if (BSRC == 0) {
#pragma unroll
            for (int i = 0; i < NB4; ++i) {
                int idx = tid + i * 256;
                int k = idx / (BN / 4), n4 = idx % (BN / 4);
                float4 v = bReg4[i];
                Bs[buf][(n4 * 4 + 0) * SA + k] = f2bf(v.x);
                Bs[buf][(n4 * 4 + 1) * SA + k] = f2bf(v.y);
                Bs[buf][(n4 * 4 + 2) * SA + k] = f2bf(v.z);
                Bs[buf][(n4 * 4 + 3) * SA + k] = f2bf(v.w);
            }
        } else {
#pragma unroll
            for (int i = 0; i < NBH; ++i) {
                int idx = tid + i * 256;
                int n = idx / (BK / 8), c8 = idx % (BK / 8);
                *(uint4*)&Bs[buf][n * SA + c8 * 8] = bRegH[i];
            }
        }
    };

    const int nk = K / BK;
    loadA(0);
    loadB(0);
    storeA(0, 0);
    storeB(0);
    __syncthreads();

    for (int kb = 0; kb < nk; ++kb) {
        const int cur = kb & 1, nxt = cur ^ 1;
        const bool more = (kb + 1 < nk);
        if (more) { loadA((kb + 1) * BK); loadB((kb + 1) * BK); }  // in flight over WMMAs

#pragma unroll
        for (int kh = 0; kh < KH; ++kh) {
            v16bf aF[WTM], bF[WTN];
#pragma unroll
            for (int tm = 0; tm < WTM; ++tm)
                aF[tm] = load_frag(&As[cur][(rowBase + tm * 16) * SA + kh * 32], SA, lane);
#pragma unroll
            for (int tn = 0; tn < WTN; ++tn)
                bF[tn] = load_frag(&Bs[cur][(colBase + tn * 16) * SA + kh * 32], SA, lane);
#pragma unroll
            for (int tm = 0; tm < WTM; ++tm)
#pragma unroll
                for (int tn = 0; tn < WTN; ++tn)
                    acc[tm][tn] = __builtin_amdgcn_wmma_f32_16x16x32_bf16(
                        false, aF[tm], false, bF[tn], (short)0, acc[tm][tn], false, false);
        }

        if (more) { storeA(nxt, (kb + 1) * BK); storeB(nxt); }
        __syncthreads();
    }

    // ---- epilogue: C layout VGPR v -> M = v + 8*(lane>=16), N = lane&15 ----
#pragma unroll
    for (int tm = 0; tm < WTM; ++tm) {
#pragma unroll
        for (int tn = 0; tn < WTN; ++tn) {
#pragma unroll
            for (int v = 0; v < 8; ++v) {
                int row = blockRow + rowBase + tm * 16 + v + 8 * (lane >> 4);
                int col = colBase + tn * 16 + (lane & 15);
                float s   = scale[row];
                float val = s * acc[tm][tn][v];
                if (EPI == 0) {
                    ((bf16_t*)outp)[(size_t)col * ldo + row] = f2bf(val);
                } else if (EPI == 1) {
                    ((bf16_t*)outp)[(size_t)row * ldo + col] = f2bf(fmaxf(val, 0.f));
                } else {
                    if (col < ncolsOut)
                        ((float*)outp)[(size_t)row * ncolsOut + col] = val;
                }
            }
        }
    }
}

// ---------------- loss kernel ----------------
__global__ __launch_bounds__(256) void loss_kernel(const float* __restrict__ logits,
                                                   const int* __restrict__ sup,
                                                   const int* __restrict__ lab,
                                                   float* __restrict__ lossOut) {
    int tid = threadIdx.x;
    float local = 0.f;
    for (int i = tid; i < GNSUP; i += 256) {
        int r = sup[i];
        const float* lg = logits + (size_t)r * GDOUT;
        float mx = lg[0];
        for (int j = 1; j < GDOUT; ++j) mx = fmaxf(mx, lg[j]);
        float se = 0.f;
        for (int j = 0; j < GDOUT; ++j) se += expf(lg[j] - mx);
        float lp = lg[lab[i]] - mx - logf(se);
        local -= lp;
    }
    __shared__ float red[256];
    red[tid] = local;
    __syncthreads();
    for (int off = 128; off > 0; off >>= 1) {
        if (tid < off) red[tid] += red[tid + off];
        __syncthreads();
    }
    if (tid == 0) lossOut[0] = red[0] / (float)GNSUP;
}

// ---------------- host side ----------------
extern "C" void kernel_launch(void* const* d_in, const int* in_sizes, int n_in,
                              void* d_out, int out_size, void* d_ws, size_t ws_size,
                              hipStream_t stream) {
    (void)in_sizes; (void)n_in; (void)out_size; (void)ws_size;

    const float* x   = (const float*)d_in[0];   // [N, DIN]
    const float* adj = (const float*)d_in[1];   // [N, N]
    const float* W1  = (const float*)d_in[2];   // [DIN, DH]
    const float* W2  = (const float*)d_in[3];   // [DH, DOUT]
    const int*   sup = (const int*)d_in[4];     // [NSUP]
    const int*   lab = (const int*)d_in[5];     // [NSUP]
    float* out = (float*)d_out;                 // logits [N*DOUT] ++ loss [1]

    char* ws = (char*)d_ws;
    float*  row_inv = (float*)(ws);
    float*  col_inv = (float*)(ws + (size_t)GN * 4);
    float*  colsq   = (float*)(ws + (size_t)GN * 8);
    bf16_t* B1t = (bf16_t*)(ws + (size_t)GN * 12);                           // [128, N] bf16 (transposed)
    bf16_t* Hb  = (bf16_t*)(ws + (size_t)GN * 12 + (size_t)GN * GDH * 2);    // [N, 128] bf16
    bf16_t* B2t = (bf16_t*)(ws + (size_t)GN * 12 + (size_t)GN * GDH * 4);    // [64, N]  bf16 (transposed)

    // 1) normalization vectors
    rownorm_kernel<<<GN, 256, 0, stream>>>(adj, row_inv);
    colsq_init_kernel<<<GN / 256, 256, 0, stream>>>(colsq);
    colsq_part_kernel<<<dim3(GN / 256, 32), 256, 0, stream>>>(adj, colsq);
    colinv_kernel<<<GN / 256, 256, 0, stream>>>(colsq, col_inv);

    // 2) B1t = (diag(col_inv) * (x @ W1))^T   -> bf16 [128, N]
    gemm_wmma_kernel<128, 128, 64, 4, 2, false, /*A f32*/0, /*B f32*/0, /*EPI*/0>
        <<<dim3(1, GN / 128), 256, 0, stream>>>(
            x, GDIN, W1, GDH, GDH, col_inv, B1t, GN, GDH, GN, GDIN);

    // 3) Hb = relu(diag(row_inv) * ((adj+I) @ B1)) -> bf16 [N, 128]
    gemm_wmma_kernel<128, 128, 64, 4, 2, true, /*A f32*/0, /*B bf16T*/1, /*EPI*/1>
        <<<dim3(1, GN / 128), 256, 0, stream>>>(
            adj, GN, B1t, GN, GDH, row_inv, Hb, GDH, GDH, GN, GN);

    // 4) B2t = (diag(col_inv) * (Hb @ W2_pad))^T -> bf16 [64, N] (rows 40..63 = 0)
    gemm_wmma_kernel<128, 64, 64, 4, 2, false, /*A bf16*/1, /*B f32*/0, /*EPI*/0>
        <<<dim3(1, GN / 128), 256, 0, stream>>>(
            Hb, GDH, W2, GDOUT, GDOUT, col_inv, B2t, GN, GDPAD, GN, GDH);

    // 5) logits = diag(row_inv) * ((adj+I) @ B2), store first 40 f32 cols
    gemm_wmma_kernel<128, 64, 64, 4, 2, true, /*A f32*/0, /*B bf16T*/1, /*EPI*/2>
        <<<dim3(1, GN / 128), 256, 0, stream>>>(
            adj, GN, B2t, GN, GDPAD, row_inv, out, GDOUT, GDOUT, GN, GN);

    // 6) supervised NLL loss
    loss_kernel<<<1, 256, 0, stream>>>(out, sup, lab, out + (size_t)GN * GDOUT);
}
